// SequenceModel_21182778704494
// MI455X (gfx1250) — matrix operand order repaired and checked by reading, not compiled, not run on hardware
//
#include <hip/hip_runtime.h>

// LSTM: B=32, T=1024, I=512, H=512, gates G=4H=2048.
// k_prep    : fp32->bf16 weights, zero h double-buffer + barrier
// k_cvt_x   : fp32->bf16 x (streaming, ~4us at 23.3 TB/s)
// k_gemm_xg : xg = x @ W_ih^T + b  (bf16 WMMA, fp32 accum). One block = 64 gate
//             cols x 256 bt rows; W_ih slice staged in LDS and shared by all
//             8 waves; A-fragments double-buffered from global.
// k_lstm_scan: persistent 64-wave scan; W_hh slice resident in LDS for all
//             1024 steps; c-state in registers; device-wide barrier per step.

#define Bb 32
#define Tt 1024
#define Ii 512
#define Hh 512
#define Gg 2048

#define LDSROW 520          // 512 + 8 pad: 260 dwords/row -> bank shift 4/row

typedef __attribute__((ext_vector_type(16))) __bf16 v16bf;
typedef __attribute__((ext_vector_type(8)))  float  v8f;

union Frag  { uint4 q[2]; v16bf v; };
union Pack8 { uint4 u; unsigned short s[8]; };

__device__ __forceinline__ unsigned short f2bf(float f) {
  unsigned u = __builtin_bit_cast(unsigned, f);
  u += 0x7FFFu + ((u >> 16) & 1u);            // round-to-nearest-even
  return (unsigned short)(u >> 16);
}
__device__ __forceinline__ float bf2f(unsigned short s) {
  unsigned u = ((unsigned)s) << 16;
  return __builtin_bit_cast(float, u);
}
__device__ __forceinline__ uint4 pack8(float4 a, float4 b) {
  uint4 r;
  r.x = (unsigned)f2bf(a.x) | ((unsigned)f2bf(a.y) << 16);
  r.y = (unsigned)f2bf(a.z) | ((unsigned)f2bf(a.w) << 16);
  r.z = (unsigned)f2bf(b.x) | ((unsigned)f2bf(b.y) << 16);
  r.w = (unsigned)f2bf(b.z) | ((unsigned)f2bf(b.w) << 16);
  return r;
}
__device__ __forceinline__ float sigmoidf_(float x) { return 1.f / (1.f + __expf(-x)); }
__device__ __forceinline__ float tanhf_(float x) {
  float e = __expf(-2.f * x);
  return (1.f - e) / (1.f + e);
}
// A fragment (16x32 bf16): two b128 at K offsets {khalf*8, 16+khalf*8}
__device__ __forceinline__ void ldA(const unsigned short* row, int kb, Frag& f) {
  f.q[0] = *(const uint4*)(row + kb);
  f.q[1] = *(const uint4*)(row + kb + 16);
}

// ---------------- prep ------------------------------------------------------------
__global__ __launch_bounds__(256) void k_prep(const float* __restrict__ wih_f,
                                              const float* __restrict__ whh_f,
                                              unsigned short* __restrict__ wih,
                                              unsigned short* __restrict__ whh,
                                              unsigned short* __restrict__ hbuf,
                                              int* __restrict__ bar) {
  int i = blockIdx.x * blockDim.x + threadIdx.x;
  if (i < Gg * Ii) wih[i] = f2bf(wih_f[i]);
  if (i < Gg * Hh) whh[i] = f2bf(whh_f[i]);
  if (i < 2 * Bb * Hh) hbuf[i] = 0;
  if (i < 4) bar[i] = 0;
}

__global__ __launch_bounds__(256) void k_cvt_x(const float* __restrict__ xf,
                                               unsigned short* __restrict__ xb,
                                               int n8) {
  int i = blockIdx.x * blockDim.x + threadIdx.x;
  if (i < n8) {
    const float4* p = (const float4*)xf + (size_t)i * 2;
    ((uint4*)xb)[i] = pack8(p[0], p[1]);
  }
}

// ---------------- xg GEMM: [B*T,I] x [I,G] -> xg[T][G][B] bf16, bias folded -------
// grid: 4096 blocks x 256 threads. block -> (ng = 64 gate cols, 256 bt rows).
// W_ih slice (64 x 512 bf16, padded rows) staged in LDS, shared by 8 waves.
// Each wave: 2 M-tiles x 4 N-tiles, A-fragments double-buffered from global.
__global__ __launch_bounds__(256) void k_gemm_xg(const unsigned short* __restrict__ xbf,
                                                 const unsigned short* __restrict__ wih,
                                                 const float* __restrict__ b_ih,
                                                 const float* __restrict__ b_hh,
                                                 unsigned short* __restrict__ xg) {
  extern __shared__ unsigned short lds_w[];     // 64 * LDSROW bf16 (~66.5 KB)

  int tid  = threadIdx.x;
  int wavein = tid >> 5;                        // 0..7
  int lane = tid & 31;
  int l16 = lane & 15, khalf = lane >> 4;
  int ng  = blockIdx.x & 31;                    // 64-gate-col group
  int mpg = blockIdx.x >> 5;                    // 0..127 : 256-row super-group
  int mp  = mpg * 8 + wavein;                   // 0..1023 : this wave's 32 bt rows

  // ---- stage W_ih[ng*64 .. +63][0..511] into padded LDS rows ----
  for (int idx = tid; idx < 64 * 64; idx += 256) {
    int row = idx >> 6, c = idx & 63;
    *(uint4*)(lds_w + row * LDSROW + c * 8) =
        *(const uint4*)(wih + (size_t)(ng * 64 + row) * Ii + c * 8);
  }
  __syncthreads();

  long btBase = (long)mp * 32;
  int b  = (int)(btBase >> 10);
  int t0 = (int)(btBase & 1023);

  v8f acc[2][4];
#pragma unroll
  for (int m = 0; m < 2; m++)
#pragma unroll
    for (int q = 0; q < 4; q++)
#pragma unroll
      for (int r = 0; r < 8; r++) acc[m][q][r] = 0.f;

  const unsigned short* ar0 = xbf + (size_t)(btBase + l16) * Ii;
  const unsigned short* ar1 = ar0 + (size_t)16 * Ii;
  const unsigned short* wrow[4];
#pragma unroll
  for (int q = 0; q < 4; q++)
    wrow[q] = lds_w + (q * 16 + l16) * LDSROW;

  Frag a[2][2];
  ldA(ar0, khalf * 8, a[0][0]);
  ldA(ar1, khalf * 8, a[0][1]);

  int p = 0;
  for (int kk = 0; kk < Ii; kk += 32) {
    int np = p ^ 1;
    if (kk + 32 < Ii) {                 // prefetch next A chunk while MMAs run
      ldA(ar0, kk + 32 + khalf * 8, a[np][0]);
      ldA(ar1, kk + 32 + khalf * 8, a[np][1]);
    }
#pragma unroll
    for (int q = 0; q < 4; q++) {
      Frag bw;                          // B from LDS: cheap ds_load_b128 pair
      const uint4* p4 = (const uint4*)(wrow[q] + kk + khalf * 8);
      bw.q[0] = p4[0];
      bw.q[1] = p4[2];
      acc[0][q] = __builtin_amdgcn_wmma_f32_16x16x32_bf16(false, a[p][0].v, false, bw.v,
                                                          (short)0, acc[0][q], false, false);
      acc[1][q] = __builtin_amdgcn_wmma_f32_16x16x32_bf16(false, a[p][1].v, false, bw.v,
                                                          (short)0, acc[1][q], false, false);
    }
    p = np;
  }

#pragma unroll
  for (int m = 0; m < 2; m++)
#pragma unroll
    for (int q = 0; q < 4; q++) {
      int g = ng * 64 + q * 16 + l16;       // C layout: lane%16 = N col
      float bias = b_ih[g] + b_hh[g];
#pragma unroll
      for (int r = 0; r < 8; r++) {
        int t = t0 + m * 16 + r + khalf * 8;   // VGPR r: M row = r + 8*(lane/16)
        xg[((size_t)t * Gg + g) * Bb + b] = f2bf(acc[m][q][r] + bias);
      }
    }
}

// ---------------- persistent recurrent scan ---------------------------------------
// 16 blocks x 128 threads = 64 waves; wave -> (mt 0..1, nt 0..31).
// Dynamic LDS: W_hh slice for this WG's 2 h-tiles (128 rows x 520, padded) loaded
// ONCE, plus h staging buffer (32 rows x 520). c-state lives in registers.
__global__ __launch_bounds__(128) void k_lstm_scan(const unsigned short* __restrict__ xg,
                                                   const unsigned short* __restrict__ whh,
                                                   unsigned short* __restrict__ hbuf,
                                                   float* __restrict__ outs,
                                                   float* __restrict__ hT,
                                                   float* __restrict__ cT,
                                                   int* __restrict__ bar) {
  extern __shared__ unsigned short smem[];
  unsigned short* lds_w = smem;                    // 128 * LDSROW
  unsigned short* lds_h = smem + 128 * LDSROW;     // 32  * LDSROW

  int tid  = threadIdx.x;
  int wave = tid >> 5, lane = tid & 31;
  int l16 = lane & 15, khalf = lane >> 4;
  int w  = blockIdx.x * 4 + wave;   // 0..63
  int mt = w & 1;                   // b-tile
  int nt = w >> 1;                  // h-column tile 0..31  (block owns nt pair)
  int nl = nt & 1;                  // local h-tile index in this WG
  int hcol = nt * 16 + l16;
  int nwg = (int)gridDim.x;

  // ---- preload W_hh slice into LDS (once; reused 1024x) ----
  // logical row = (nl*4 + q)*16 + r  -> gate col g = q*512 + nt*16 + r
  for (int idx = tid; idx < 128 * 64; idx += 128) {
    int row = idx >> 6, c = idx & 63;
    int lnl = row >> 6, q = (row >> 4) & 3, r = row & 15;
    int g = q * Hh + (blockIdx.x * 2 + lnl) * 16 + r;
    *(uint4*)(lds_w + row * LDSROW + c * 8) =
        *(const uint4*)(whh + (size_t)g * Hh + c * 8);
  }
  __syncthreads();

  const unsigned short* wrow[4];
#pragma unroll
  for (int q = 0; q < 4; q++)
    wrow[q] = lds_w + ((nl * 4 + q) * 16 + l16) * LDSROW;
  const unsigned short* hrow = lds_h + (mt * 16 + l16) * LDSROW;

  v8f cst;
#pragma unroll
  for (int r = 0; r < 8; r++) cst[r] = 0.f;
  float hv[8];
#pragma unroll
  for (int r = 0; r < 8; r++) hv[r] = 0.f;
  int phase = 0;

  for (int t = 0; t < Tt; ++t) {
    const unsigned short* hcur = hbuf + (size_t)(t & 1) * (Bb * Hh);
    unsigned short* hnext      = hbuf + (size_t)((t + 1) & 1) * (Bb * Hh);

    // issue xg[t] tile loads first: latency hides under staging + syncthreads
    uint4 xgl[4];
    const unsigned short* xgt = xg + (size_t)t * Gg * Bb;
#pragma unroll
    for (int q = 0; q < 4; q++)
      xgl[q] = *(const uint4*)(xgt + (size_t)(q * Hh + nt * 16 + l16) * Bb +
                               mt * 16 + khalf * 8);

    // stage h_t into padded LDS rows
    for (int idx = tid; idx < 32 * 64; idx += 128) {
      int row = idx >> 6, c = idx & 63;
      *(uint4*)(lds_h + row * LDSROW + c * 8) =
          *(const uint4*)(hcur + row * Hh + c * 8);
    }
    __syncthreads();

    // init gate accumulators from xg[t]
    v8f acc[4];
#pragma unroll
    for (int q = 0; q < 4; q++) {
      Pack8 pk; pk.u = xgl[q];
#pragma unroll
      for (int r = 0; r < 8; r++) acc[q][r] = bf2f(pk.s[r]);
    }

    // gates += h_t @ W_hh^T  : all fragments from LDS, double-buffered
    Frag a[2], bw[2][4];
    a[0].q[0] = *(const uint4*)(hrow + khalf * 8);
    a[0].q[1] = *(const uint4*)(hrow + khalf * 8 + 16);
#pragma unroll
    for (int q = 0; q < 4; q++) {
      const uint4* p4 = (const uint4*)(wrow[q] + khalf * 8);
      bw[0][q].q[0] = p4[0];
      bw[0][q].q[1] = p4[2];
    }
    int p = 0;
    for (int kk = 0; kk < Hh; kk += 32) {
      int np = p ^ 1;
      if (kk + 32 < Hh) {
        a[np].q[0] = *(const uint4*)(hrow + kk + 32 + khalf * 8);
        a[np].q[1] = *(const uint4*)(hrow + kk + 32 + khalf * 8 + 16);
#pragma unroll
        for (int q = 0; q < 4; q++) {
          const uint4* p4 = (const uint4*)(wrow[q] + kk + 32 + khalf * 8);
          bw[np][q].q[0] = p4[0];
          bw[np][q].q[1] = p4[2];
        }
      }
#pragma unroll
      for (int q = 0; q < 4; q++)
        acc[q] = __builtin_amdgcn_wmma_f32_16x16x32_bf16(false, a[p].v, false, bw[p][q].v,
                                                         (short)0, acc[q], false, false);
      p = np;
    }

    // LSTM cell (C layout: row b = mt*16 + r + 8*khalf, col = hcol)
#pragma unroll
    for (int r = 0; r < 8; r++) {
      float ig = sigmoidf_(acc[0][r]);
      float fg = sigmoidf_(acc[1][r]);
      float gg = tanhf_(acc[2][r]);
      float og = sigmoidf_(acc[3][r]);
      float cn = fg * cst[r] + ig * gg;
      cst[r] = cn;
      hv[r]  = og * tanhf_(cn);
    }
#pragma unroll
    for (int r = 0; r < 8; r++) {
      int b = mt * 16 + khalf * 8 + r;
      hnext[(size_t)b * Hh + hcol] = f2bf(hv[r]);
      outs[((size_t)b * Tt + t) * Hh + hcol] = hv[r];
    }

    // device-wide barrier between timesteps
    __threadfence();
    __syncthreads();
    if (tid == 0) {
      int ticket = __hip_atomic_fetch_add(&bar[0], 1, __ATOMIC_ACQ_REL,
                                          __HIP_MEMORY_SCOPE_AGENT);
      if (ticket == nwg - 1) {
        __hip_atomic_store(&bar[0], 0, __ATOMIC_RELAXED, __HIP_MEMORY_SCOPE_AGENT);
        __hip_atomic_fetch_add(&bar[1], 1, __ATOMIC_RELEASE, __HIP_MEMORY_SCOPE_AGENT);
      } else {
        while (__hip_atomic_load(&bar[1], __ATOMIC_ACQUIRE,
                                 __HIP_MEMORY_SCOPE_AGENT) <= phase)
          __builtin_amdgcn_s_sleep(2);
      }
    }
    phase++;
    __syncthreads();
    __threadfence();
  }

  // final (h_T, c_T)
#pragma unroll
  for (int r = 0; r < 8; r++) {
    int b = mt * 16 + khalf * 8 + r;
    hT[(size_t)b * Hh + hcol] = hv[r];
    cT[(size_t)b * Hh + hcol] = cst[r];
  }
}

// ----------------------------------------------------------------------------------
extern "C" void kernel_launch(void* const* d_in, const int* in_sizes, int n_in,
                              void* d_out, int out_size, void* d_ws, size_t ws_size,
                              hipStream_t stream) {
  (void)in_sizes; (void)n_in; (void)out_size; (void)ws_size;
  const float* x    = (const float*)d_in[0];
  const float* W_ih = (const float*)d_in[1];
  const float* W_hh = (const float*)d_in[2];
  const float* b_ih = (const float*)d_in[3];
  const float* b_hh = (const float*)d_in[4];

  float* outs = (float*)d_out;                 // [B][T][H]
  float* hT   = outs + (size_t)Bb * Tt * Hh;   // [B][H]
  float* cT   = hT + (size_t)Bb * Hh;          // [B][H]

  // workspace layout (~173 MB)
  char* ws = (char*)d_ws;
  unsigned short* xg  = (unsigned short*)ws;                        // T*G*B bf16 (134MB)
  unsigned short* xbf = xg + (size_t)Tt * Gg * Bb;                  // B*T*I bf16 (34MB)
  unsigned short* wih = xbf + (size_t)Bb * Tt * Ii;                 // 2MB
  unsigned short* whh = wih + (size_t)Gg * Ii;                      // 2MB
  unsigned short* hbuf = whh + (size_t)Gg * Hh;                     // 2 x [B][H] bf16
  int* bar = (int*)(hbuf + 2 * Bb * Hh);

  int n8 = (Bb * Tt * Ii) / 8;
  k_prep<<<(Gg * Ii + 255) / 256, 256, 0, stream>>>(W_ih, W_hh, wih, whh, hbuf, bar);
  k_cvt_x<<<(n8 + 255) / 256, 256, 0, stream>>>(x, xbf, n8);
  size_t lds_gemm = (size_t)64 * LDSROW * sizeof(unsigned short);   // ~66.5 KB
  k_gemm_xg<<<4096, 256, lds_gemm, stream>>>(xbf, wih, b_ih, b_hh, xg);
  size_t lds_scan = (size_t)(128 + 32) * LDSROW * sizeof(unsigned short); // ~163 KB
  k_lstm_scan<<<16, 128, lds_scan, stream>>>(xg, whh, hbuf, outs, hT, cT, bar);
}
